// ImageAutoregressiveRNN_40252433498470
// MI455X (gfx1250) — matrix operand order repaired and checked by reading, not compiled, not run on hardware
//
#include <hip/hip_runtime.h>
#include <hip/hip_bf16.h>

typedef _Float16 v8h  __attribute__((ext_vector_type(8)));
typedef _Float16 v16h __attribute__((ext_vector_type(16)));
typedef float    v8f  __attribute__((ext_vector_type(8)));

// ---------------------------------------------------------------------------
// Implicit-GEMM convolution + fused BatchNorm (+ residual) (+ ReLU).
// GEMM view:  M = Cout,  N = B*OH*OW,  K = Cin*KH*KW.
// Block tile: 64(M) x 128(N), K-step 32, double-buffered LDS.
// 256 threads = 8 wave32; wave (mt,nt) owns FOUR 16x16 f32 accumulators
// (2 M-subtiles x 2 N-subtiles) -> 4 v_wmma_f32_16x16x32_f16 per K-step,
// with both A fragments and both B fragments reused twice.
// Masking is specialized per shape:
//   7x7 (conv1, K=147): K-tail mask + spatial pad mask
//   3x3 (pad 1):        spatial pad mask only (K % 32 == 0)
//   1x1 (pad 0):        no masks at all (addresses only clamped)
// Masks are float multipliers (cndmask+mul), never packed bools.
// ---------------------------------------------------------------------------
#define BM 64
#define BN 128
#define BK 32

template <int KH, int KW, bool KTAIL>
__global__ __launch_bounds__(256) void conv_bn_wmma_kernel(
    const float* __restrict__ in, const float* __restrict__ wgt,
    const float* __restrict__ bng, const float* __restrict__ bnb,
    const float* __restrict__ bnm, const float* __restrict__ bnv,
    const float* __restrict__ res, float* __restrict__ out,
    int B, int Cin, int H, int W, int Cout,
    int stride, int pad, int OH, int OW, int do_relu)
{
  constexpr int  KHW     = KH * KW;
  constexpr bool SPATIAL = (KH != 1);   // 1x1 convs here are pad=0: always in-bounds
  const int Ng = B * OH * OW;
  const int Kg = Cin * KHW;

  __shared__ alignas(64) _Float16 As[2][BM][BK];   // weights  [m][k]
  __shared__ alignas(64) _Float16 Bs[2][BN][BK];   // im2col   [n][k]

  const int tid  = threadIdx.x;
  const int lane = tid & 31;
  const int wv   = tid >> 5;
  const int mt   = wv >> 2;     // 0..1 : which 32-row M half
  const int nt   = wv & 3;      // 0..3 : which 32-col N quarter

  const int m0 = blockIdx.y * BM;
  const int n0 = blockIdx.x * BN;

  // Staging: A = 64x32 (8 per thread), B = 128x32 (16 per thread).
  const int s_rowa = tid >> 2;          // 0..63
  const int s_k0a  = (tid & 3) * 8;     // 0,8,16,24
  const int s_rowb = tid >> 1;          // 0..127
  const int s_k0b  = (tid & 1) * 16;    // 0,16

  // Loop-invariant decode of this thread's output pixel for the B gather.
  // Address is clamped for out-of-range N columns; their (garbage) results
  // land in WMMA columns that are never stored.
  int ngc = n0 + s_rowb;
  if (ngc >= Ng) ngc = Ng - 1;
  const int fb  = ngc / (OH * OW);
  const int rr  = ngc % (OH * OW);
  const int foh = rr / OW;
  const int fow = rr % OW;
  const int ih_base = foh * stride - pad;
  const int iw_base = fow * stride - pad;
  const size_t in_batch_off = (size_t)fb * Cin * H * W;
  const size_t a_row_off    = (size_t)(m0 + s_rowa) * Kg;

  float aReg[8], bReg[16];
  float aMul[8], bMul[16];

#define LOAD_TILES(KB)                                                        \
  do {                                                                        \
    _Pragma("unroll")                                                         \
    for (int j = 0; j < 8; ++j) {                                             \
      const int gk = (KB) + s_k0a + j;                                        \
      if constexpr (KTAIL) {                                                  \
        const bool av = (gk < Kg);                                            \
        aMul[j] = av ? 1.f : 0.f;                                             \
        aReg[j] = wgt[av ? (a_row_off + gk) : (size_t)0];                     \
      } else {                                                                \
        aReg[j] = wgt[a_row_off + gk];                                        \
      }                                                                       \
    }                                                                         \
    _Pragma("unroll")                                                         \
    for (int j = 0; j < 16; ++j) {                                            \
      const int gk = (KB) + s_k0b + j;                                        \
      const int ci = gk / KHW;                                                \
      const int r2 = gk - ci * KHW;                                           \
      const int kh = r2 / KW;                                                 \
      const int kw = r2 - kh * KW;                                            \
      const int ih = ih_base + kh;                                            \
      const int iw = iw_base + kw;                                            \
      if constexpr (SPATIAL) {                                                \
        bool bv = ((unsigned)ih < (unsigned)H) &&                             \
                  ((unsigned)iw < (unsigned)W);                               \
        if constexpr (KTAIL) bv = bv && (gk < Kg);                            \
        bMul[j] = bv ? 1.f : 0.f;                                             \
        bReg[j] =                                                             \
            in[bv ? (in_batch_off + ((size_t)ci * H + ih) * W + iw)           \
                  : (size_t)0];                                               \
      } else {                                                                \
        bReg[j] = in[in_batch_off + ((size_t)ci * H + ih) * W + iw];          \
      }                                                                       \
    }                                                                         \
  } while (0)

#define STORE_TILES(BUF)                                                      \
  do {                                                                        \
    _Pragma("unroll")                                                         \
    for (int j = 0; j < 8; ++j) {                                             \
      if constexpr (KTAIL)                                                    \
        As[BUF][s_rowa][s_k0a + j] = (_Float16)(aReg[j] * aMul[j]);           \
      else                                                                    \
        As[BUF][s_rowa][s_k0a + j] = (_Float16)aReg[j];                       \
    }                                                                         \
    _Pragma("unroll")                                                         \
    for (int j = 0; j < 16; ++j) {                                            \
      if constexpr (SPATIAL)                                                  \
        Bs[BUF][s_rowb][s_k0b + j] = (_Float16)(bReg[j] * bMul[j]);           \
      else                                                                    \
        Bs[BUF][s_rowb][s_k0b + j] = (_Float16)bReg[j];                       \
    }                                                                         \
  } while (0)

  v8f acc00 = {}, acc01 = {}, acc10 = {}, acc11 = {};

  // Prologue: stage tile 0 into buffer 0.
  LOAD_TILES(0);
  STORE_TILES(0);
  __syncthreads();

  const int lm  = lane & 15;
  const int akb = (lane < 16) ? 0 : 16;   // A frag K-base (contiguous)
  const int bkb = (lane < 16) ? 0 : 16;   // B frag K-base (contiguous)

  int buf = 0;
  for (int kb = 0; kb < Kg; kb += BK) {
    const bool hasNext = (kb + BK) < Kg;
    if (hasNext) LOAD_TILES(kb + BK);     // global loads in flight during WMMA

    // B fragments: lane<16 -> K 0..15, lane>=16 -> K 16..31, N = lane&15.
    const v16h b0 = *(const v16h*)&Bs[buf][nt * 32 + lm][bkb];
    const v16h b1 = *(const v16h*)&Bs[buf][nt * 32 + 16 + lm][bkb];

    // A fragments: lane<16 holds K 0..15, lane>=16 K 16..31.
    const v8h lo0 = *(const v8h*)&As[buf][mt * 32 + lm][akb];
    const v8h hi0 = *(const v8h*)&As[buf][mt * 32 + lm][akb + 8];
    const v8h lo1 = *(const v8h*)&As[buf][mt * 32 + 16 + lm][akb];
    const v8h hi1 = *(const v8h*)&As[buf][mt * 32 + 16 + lm][akb + 8];
    v16h a0, a1;
    #pragma unroll
    for (int i = 0; i < 8; ++i) {
      a0[i] = lo0[i]; a0[i + 8] = hi0[i];
      a1[i] = lo1[i]; a1[i + 8] = hi1[i];
    }

    acc00 = __builtin_amdgcn_wmma_f32_16x16x32_f16(
        false, a0, false, b0, (short)0, acc00, false, false);
    acc01 = __builtin_amdgcn_wmma_f32_16x16x32_f16(
        false, a0, false, b1, (short)0, acc01, false, false);
    acc10 = __builtin_amdgcn_wmma_f32_16x16x32_f16(
        false, a1, false, b0, (short)0, acc10, false, false);
    acc11 = __builtin_amdgcn_wmma_f32_16x16x32_f16(
        false, a1, false, b1, (short)0, acc11, false, false);

    if (hasNext) STORE_TILES(buf ^ 1);    // fill the other buffer
    __syncthreads();
    buf ^= 1;
  }
#undef LOAD_TILES
#undef STORE_TILES

  // ---- epilogue: BN (+residual) (+ReLU), NCHW store ----
  // Cout is always a multiple of 64 here, so no M guard is needed.
  const size_t chan_stride = (size_t)OH * OW;
  const int ng_base = n0 + nt * 32 + lm;
  #pragma unroll
  for (int nh = 0; nh < 2; ++nh) {
    const int ng = ng_base + nh * 16;
    if (ng < Ng) {
      const int ob  = ng / (OH * OW);
      const int r   = ng % (OH * OW);
      const int ooh = r / OW;
      const int oow = r % OW;
      const size_t pix_off = (((size_t)ob * Cout) * OH + ooh) * OW + oow;
      #pragma unroll
      for (int mh = 0; mh < 2; ++mh) {
        const v8f acc = nh ? (mh ? acc11 : acc01) : (mh ? acc10 : acc00);
        #pragma unroll
        for (int rI = 0; rI < 8; ++rI) {
          const int gm = m0 + mt * 32 + mh * 16 + rI + ((lane < 16) ? 0 : 8);
          const float scale = bng[gm] * rsqrtf(bnv[gm] + 1e-5f);
          const float bias  = bnb[gm] - bnm[gm] * scale;
          float v = acc[rI] * scale + bias;
          const size_t oidx = pix_off + (size_t)gm * chan_stride;
          if (res) v += res[oidx];
          if (do_relu) v = fmaxf(v, 0.f);
          out[oidx] = v;
        }
      }
    }
  }
}

// ---------------------------------------------------------------------------
// MaxPool 3x3 stride 2 pad 1 (NCHW).
// ---------------------------------------------------------------------------
__global__ __launch_bounds__(256) void maxpool_kernel(
    const float* __restrict__ in, float* __restrict__ out,
    int B, int C, int H, int W, int OH, int OW)
{
  const int idx = blockIdx.x * blockDim.x + threadIdx.x;
  const int total = B * C * OH * OW;
  if (idx >= total) return;
  int t = idx;
  const int ow = t % OW; t /= OW;
  const int oh = t % OH; t /= OH;
  const int c  = t % C;  t /= C;
  const int b  = t;
  float m = -3.402823466e+38f;
  #pragma unroll
  for (int kh = 0; kh < 3; ++kh) {
    const int ih = oh * 2 - 1 + kh;
    if (ih < 0 || ih >= H) continue;
    #pragma unroll
    for (int kw = 0; kw < 3; ++kw) {
      const int iw = ow * 2 - 1 + kw;
      if (iw < 0 || iw >= W) continue;
      m = fmaxf(m, in[(((size_t)b * C + c) * H + ih) * W + iw]);
    }
  }
  out[idx] = m;
}

// ---------------------------------------------------------------------------
// Global average pool: (B, C, HW) -> (B, C)
// ---------------------------------------------------------------------------
__global__ __launch_bounds__(256) void avgpool_kernel(
    const float* __restrict__ in, float* __restrict__ out, int B, int C, int HW)
{
  const int idx = blockIdx.x * blockDim.x + threadIdx.x;
  if (idx >= B * C) return;
  const float* p = in + (size_t)idx * HW;
  float s = 0.f;
  for (int i = 0; i < HW; ++i) s += p[i];
  out[idx] = s / (float)HW;
}

// ---------------------------------------------------------------------------
// Head: FC(512->128) + embedding concat (->152) + fusion MLP (->64->64).
// One block per batch row.
// ---------------------------------------------------------------------------
__global__ __launch_bounds__(128) void head_kernel(
    const float* __restrict__ pooled, const float* __restrict__ fc_w,
    const float* __restrict__ fc_b, const float* __restrict__ cemb,
    const float* __restrict__ oemb, const int* __restrict__ cidx,
    const int* __restrict__ oidx, const float* __restrict__ w1,
    const float* __restrict__ b1, const float* __restrict__ w2,
    const float* __restrict__ b2, float* __restrict__ ctx)
{
  const int b = blockIdx.x;
  const int t = threadIdx.x;
  __shared__ float raw[152];
  __shared__ float h1[64];

  {
    float s = fc_b[t];
    const float* prow = pooled + (size_t)b * 512;
    const float* wrow = fc_w + (size_t)t * 512;
    for (int k = 0; k < 512; ++k) s += prow[k] * wrow[k];
    raw[t] = s;
  }
  if (t < 16) raw[128 + t] = cemb[cidx[b] * 16 + t];
  if (t < 8)  raw[144 + t] = oemb[oidx[b] * 8 + t];
  __syncthreads();

  if (t < 64) {
    float s = b1[t];
    const float* wrow = w1 + (size_t)t * 152;
    for (int k = 0; k < 152; ++k) s += raw[k] * wrow[k];
    h1[t] = fmaxf(s, 0.f);
  }
  __syncthreads();

  if (t < 64) {
    float s = b2[t];
    const float* wrow = w2 + (size_t)t * 64;
    for (int k = 0; k < 64; ++k) s += h1[k] * wrow[k];
    ctx[(size_t)b * 64 + t] = fmaxf(s, 0.f);
  }
}

// ---------------------------------------------------------------------------
// Autoregressive GRU rollout, single block. x_t = [ctx(64), prev_pred(1)].
// Writes preds (B, T) straight to d_out.
// ---------------------------------------------------------------------------
__global__ __launch_bounds__(256) void gru_kernel(
    const float* __restrict__ ctx, const float* __restrict__ wih,
    const float* __restrict__ whh, const float* __restrict__ bih,
    const float* __restrict__ bhh, const float* __restrict__ ow,
    const float* __restrict__ ob, const int* __restrict__ seqlen_p,
    float* __restrict__ out)
{
  __shared__ float cs[32][64];
  __shared__ float hs[32][64];
  __shared__ float hn[32][64];
  __shared__ float prev[32];
  __shared__ int Ts;

  const int t = threadIdx.x;
  for (int i = t; i < 32 * 64; i += 256) {
    cs[i / 64][i % 64] = ctx[i];
    hs[i / 64][i % 64] = 0.f;
  }
  if (t < 32) prev[t] = -1.0f;
  if (t == 0) Ts = seqlen_p[0];
  __syncthreads();
  const int T = Ts;

  for (int step = 0; step < T; ++step) {
    for (int e = t; e < 32 * 64; e += 256) {
      const int b = e >> 6;
      const int u = e & 63;
      float gir = bih[u],       giz = bih[64 + u],  gin_ = bih[128 + u];
      float ghr = bhh[u],       ghz = bhh[64 + u],  ghn  = bhh[128 + u];
      for (int k = 0; k < 64; ++k) {
        const float xv = cs[b][k];
        gir  += xv * wih[u * 65 + k];
        giz  += xv * wih[(64 + u) * 65 + k];
        gin_ += xv * wih[(128 + u) * 65 + k];
        const float hv = hs[b][k];
        ghr += hv * whh[u * 64 + k];
        ghz += hv * whh[(64 + u) * 64 + k];
        ghn += hv * whh[(128 + u) * 64 + k];
      }
      const float pv = prev[b];
      gir  += pv * wih[u * 65 + 64];
      giz  += pv * wih[(64 + u) * 65 + 64];
      gin_ += pv * wih[(128 + u) * 65 + 64];
      const float r = 1.f / (1.f + __expf(-(gir + ghr)));
      const float z = 1.f / (1.f + __expf(-(giz + ghz)));
      const float n = tanhf(gin_ + r * ghn);
      hn[b][u] = (1.f - z) * n + z * hs[b][u];
    }
    __syncthreads();
    for (int e = t; e < 32 * 64; e += 256) hs[e >> 6][e & 63] = hn[e >> 6][e & 63];
    __syncthreads();
    if (t < 32) {
      float s = ob[0];
      for (int k = 0; k < 64; ++k) s += hs[t][k] * ow[k];
      prev[t] = s;
      out[t * T + step] = s;            // (B, T, 1) layout
    }
    __syncthreads();
  }
}

// ---------------------------------------------------------------------------
// Host side. Param pointer indices follow jax pytree flattening of the params
// dict (sorted dict keys; tuples/lists in order):
//  0 images  1 country_idx  2 os_idx  3 seq_len
//  4..7  bn1(g,b,m,v)   8 conv1   9 country_emb  10 fc_b  11 fc_w
// 12 fus_b1 13 fus_b2 14 fus_w1 15 fus_w2
// 16 gru_bhh 17 gru_bih 18 gru_whh 19 gru_wih
// layers:
//  L0B0: 20..23 bn1, 24..27 bn2, 28 w1, 29 w2
//  L0B1: 30..33 bn1, 34..37 bn2, 38 w1, 39 w2
//  L1B0: 40..43 bn1, 44..47 bn2, 48..51 bnd, 52 w1, 53 w2, 54 wd
//  L1B1: 55..58 bn1, 59..62 bn2, 63 w1, 64 w2
//  L2B0: 65..68 bn1, 69..72 bn2, 73..76 bnd, 77 w1, 78 w2, 79 wd
//  L2B1: 80..83 bn1, 84..87 bn2, 88 w1, 89 w2
//  L3B0: 90..93 bn1, 94..97 bn2, 98..101 bnd, 102 w1, 103 w2, 104 wd
//  L3B1: 105..108 bn1, 109..112 bn2, 113 w1, 114 w2
// 115 os_emb  116 out_b  117 out_w
// ---------------------------------------------------------------------------
extern "C" void kernel_launch(void* const* d_in, const int* in_sizes, int n_in,
                              void* d_out, int out_size, void* d_ws, size_t ws_size,
                              hipStream_t stream) {
  (void)in_sizes; (void)n_in; (void)out_size; (void)ws_size;

  auto F = [&](int i) -> const float* { return (const float*)d_in[i]; };

  // Workspace carve-up (floats).
  float* ws = (float*)d_ws;
  size_t o = 0;
  float* bufC1 = ws + o; o += 25690112;  // conv1 out: 32*64*112*112
  float* bufP  = ws + o; o += 6422528;   // 32*64*56*56 rotation buffers
  float* bufA  = ws + o; o += 6422528;
  float* bufB  = ws + o; o += 6422528;
  float* pooled = ws + o; o += 32 * 512;
  float* ctx    = ws + o; o += 32 * 64;

  auto conv = [&](const float* in, int pw, int pbn, const float* res, float* out,
                  int B, int Cin, int H, int W, int Cout, int KH, int KW,
                  int s, int p, int relu) {
    const int OH = (H + 2 * p - KH) / s + 1;
    const int OW = (W + 2 * p - KW) / s + 1;
    const int Ng = B * OH * OW;
    dim3 grid((Ng + BN - 1) / BN, (Cout + BM - 1) / BM);
    if (KH == 7) {
      conv_bn_wmma_kernel<7, 7, true><<<grid, 256, 0, stream>>>(
          in, F(pw), F(pbn), F(pbn + 1), F(pbn + 2), F(pbn + 3), res, out,
          B, Cin, H, W, Cout, s, p, OH, OW, relu);
    } else if (KH == 3) {
      conv_bn_wmma_kernel<3, 3, false><<<grid, 256, 0, stream>>>(
          in, F(pw), F(pbn), F(pbn + 1), F(pbn + 2), F(pbn + 3), res, out,
          B, Cin, H, W, Cout, s, p, OH, OW, relu);
    } else {
      conv_bn_wmma_kernel<1, 1, false><<<grid, 256, 0, stream>>>(
          in, F(pw), F(pbn), F(pbn + 1), F(pbn + 2), F(pbn + 3), res, out,
          B, Cin, H, W, Cout, s, p, OH, OW, relu);
    }
  };

  const float* images = F(0);

  // Stem
  conv(images, 8, 4, nullptr, bufC1, 32, 3, 224, 224, 64, 7, 7, 2, 3, 1);
  {
    const int total = 32 * 64 * 56 * 56;
    maxpool_kernel<<<(total + 255) / 256, 256, 0, stream>>>(
        bufC1, bufP, 32, 64, 112, 112, 56, 56);
  }

  // Layer 1 (64ch, 56x56)
  conv(bufP, 28, 20, nullptr, bufA, 32, 64, 56, 56, 64, 3, 3, 1, 1, 1);
  conv(bufA, 29, 24, bufP,    bufB, 32, 64, 56, 56, 64, 3, 3, 1, 1, 1);
  conv(bufB, 38, 30, nullptr, bufA, 32, 64, 56, 56, 64, 3, 3, 1, 1, 1);
  conv(bufA, 39, 34, bufB,    bufP, 32, 64, 56, 56, 64, 3, 3, 1, 1, 1);

  // Layer 2 (128ch, 28x28), downsample block first
  conv(bufP, 54, 48, nullptr, bufB, 32, 64, 56, 56, 128, 1, 1, 2, 0, 0);  // wd -> idn
  conv(bufP, 52, 40, nullptr, bufA, 32, 64, 56, 56, 128, 3, 3, 2, 1, 1);
  conv(bufA, 53, 44, bufB,    bufP, 32, 128, 28, 28, 128, 3, 3, 1, 1, 1);
  conv(bufP, 63, 55, nullptr, bufA, 32, 128, 28, 28, 128, 3, 3, 1, 1, 1);
  conv(bufA, 64, 59, bufP,    bufB, 32, 128, 28, 28, 128, 3, 3, 1, 1, 1);

  // Layer 3 (256ch, 14x14)
  conv(bufB, 79, 73, nullptr, bufP, 32, 128, 28, 28, 256, 1, 1, 2, 0, 0);  // wd -> idn
  conv(bufB, 77, 65, nullptr, bufA, 32, 128, 28, 28, 256, 3, 3, 2, 1, 1);
  conv(bufA, 78, 69, bufP,    bufB, 32, 256, 14, 14, 256, 3, 3, 1, 1, 1);
  conv(bufB, 88, 80, nullptr, bufA, 32, 256, 14, 14, 256, 3, 3, 1, 1, 1);
  conv(bufA, 89, 84, bufB,    bufP, 32, 256, 14, 14, 256, 3, 3, 1, 1, 1);

  // Layer 4 (512ch, 7x7)
  conv(bufP, 104, 98, nullptr, bufB, 32, 256, 14, 14, 512, 1, 1, 2, 0, 0); // wd -> idn
  conv(bufP, 102, 90, nullptr, bufA, 32, 256, 14, 14, 512, 3, 3, 2, 1, 1);
  conv(bufA, 103, 94, bufB,    bufP, 32, 512, 7, 7,  512, 3, 3, 1, 1, 1);
  conv(bufP, 113, 105, nullptr, bufA, 32, 512, 7, 7, 512, 3, 3, 1, 1, 1);
  conv(bufA, 114, 109, bufP,   bufB, 32, 512, 7, 7,  512, 3, 3, 1, 1, 1);

  // Head
  avgpool_kernel<<<(32 * 512 + 255) / 256, 256, 0, stream>>>(bufB, pooled, 32, 512, 49);
  head_kernel<<<32, 128, 0, stream>>>(
      pooled, F(11), F(10), F(9), F(115),
      (const int*)d_in[1], (const int*)d_in[2],
      F(14), F(12), F(15), F(13), ctx);
  gru_kernel<<<1, 256, 0, stream>>>(
      ctx, F(19), F(18), F(17), F(16), F(117), F(116),
      (const int*)d_in[3], (float*)d_out);
}
